// MECIModel_704374637088
// MI455X (gfx1250) — matrix operand order repaired and checked by reading, not compiled, not run on hardware
//
#include <hip/hip_runtime.h>

#define H    768
#define TWOH 1536
#define Ln   4096
#define Pn   8192
#define KP   8
#define EVn  2048
#define KE   6
#define En   16384
#define NR   32

#define MT    64     // M rows per block (4 waves x 16)
#define NSUB  8      // 8 x 16 = 128 N columns per stripe
#define NBN   6      // 6 stripes -> full N = 768
#define KSTEPS 48    // 1536 / 32
#define LDB   40     // padded B-tile row pitch in bf16 (80 B -> conflict-free frag reads)

typedef __bf16 bf16;
typedef __attribute__((ext_vector_type(16))) __bf16    v16bf;
typedef __attribute__((ext_vector_type(8)))  float     v8f;
typedef __attribute__((ext_vector_type(4)))  int       v4i;
typedef __attribute__((ext_vector_type(8)))  int       v8i;
typedef __attribute__((ext_vector_type(4)))  unsigned  v4u;

union FragAB { v16bf v; uint4 q[2]; };

// ---- CDNA5 data-mover paths (guarded; fall back gracefully) ----
#if defined(__has_builtin)
#if __has_builtin(__builtin_amdgcn_global_load_async_to_lds_b128)
#define HAVE_ASYNC_LDS 1
#endif
#if __has_builtin(__builtin_amdgcn_tensor_load_to_lds)
#define HAVE_TDM 1
#endif
#endif

typedef __attribute__((address_space(1))) v4i* as1_v4i;          // global int4*
typedef __attribute__((address_space(3))) v4i* as3_v4i;          // LDS int4*
typedef __attribute__((address_space(3))) const void* as3_cvptr;

__device__ __forceinline__ unsigned lds_offset(const void* p) {
  return (unsigned)(uintptr_t)(as3_cvptr)p;   // LDS byte offset of a __shared__ object
}

__device__ __forceinline__ void async_copy16(const void* gsrc, void* ldst) {
#if defined(HAVE_ASYNC_LDS)
  __builtin_amdgcn_global_load_async_to_lds_b128((as1_v4i)gsrc, (as3_v4i)ldst, 0, 0);
#else
  *(uint4*)ldst = *(const uint4*)gsrc;
#endif
}

#if defined(HAVE_TDM)
// TDM 2D tile: 128 rows x 64 B, row stride 3072 B, LDS pad 16 B per 64 B row (-> 80 B pitch).
// D# per CDNA5 ISA 8.3/8.4: data_size=8B units; tensor==tile; groups 2/3 zero (2D).
// 6-arg toolchain variant: (uint32x4 g0, int32x8 g1, int32x4 g2, int32x4 g3, int32x8, cpol).
__device__ __forceinline__ void tdm_load_tile_128x64B(const void* gsrc, unsigned ldsoff) {
  const unsigned long long ga = (unsigned long long)(uintptr_t)gsrc;
  v4u g0;
  g0.x = 1u;                                                    // count=1: valid user D#
  g0.y = ldsoff;                                                // lds_addr (bytes)
  g0.z = (unsigned)(ga & 0xffffffffull);                        // global_addr[31:0]
  g0.w = (unsigned)((ga >> 32) & 0x1ffffffull) | (2u << 30);    // addr[56:32] | type=2
  v8i g1;
  g1[0] = (3 << 16)      // data_size = 8 B
        | (1 << 20)      // pad_enable
        | (3 << 22)      // pad_interval: 16 DWORDs (64 B)
        | (3 << 25);     // pad_amount: 4 DWORDs (16 B)
  g1[1] = (8 << 16);     // tensor_dim0[15:0] = 8 (8B units = 64 B row)
  g1[2] = (128 << 16);   // tensor_dim1[15:0] = 128 rows
  g1[3] = (8 << 16);     // tile_dim0 = 8 (8B units = 64 B)
  g1[4] = 128;           // tile_dim1 = 128, tile_dim2 = 0
  g1[5] = 384;           // tensor_dim0_stride = 3072 B / 8
  g1[6] = 0;
  g1[7] = 0;
  const v4i gz4 = {0, 0, 0, 0};
  const v8i gz8 = {0, 0, 0, 0, 0, 0, 0, 0};
  __builtin_amdgcn_tensor_load_to_lds(g0, g1, gz4, gz4, gz8, 0);
}
#endif

// wait until only the just-issued prefetch stage (if any) is still in flight
__device__ __forceinline__ void wait_stage(bool prefetch_inflight) {
#if defined(HAVE_TDM)
#if __has_builtin(__builtin_amdgcn_s_wait_tensorcnt)
  if (prefetch_inflight) __builtin_amdgcn_s_wait_tensorcnt(2);
  else                   __builtin_amdgcn_s_wait_tensorcnt(0);
#else
  if (prefetch_inflight) asm volatile("s_wait_tensorcnt 0x2" ::: "memory");
  else                   asm volatile("s_wait_tensorcnt 0x0" ::: "memory");
#endif
#elif defined(HAVE_ASYNC_LDS)
#if __has_builtin(__builtin_amdgcn_s_wait_asynccnt)
  if (prefetch_inflight) __builtin_amdgcn_s_wait_asynccnt(8);
  else                   __builtin_amdgcn_s_wait_asynccnt(0);
#else
  if (prefetch_inflight) asm volatile("s_wait_asynccnt 0x8" ::: "memory");
  else                   asm volatile("s_wait_asynccnt 0x0" ::: "memory");
#endif
#endif
}

__device__ __forceinline__ void split_bf16(float x, bf16& hi, bf16& lo) {
  hi = (bf16)x;
  lo = (bf16)(x - (float)hi);
}

// ---- event masked-mean pooling -> bf16 hi/lo rows (2048 x 768) ----
__global__ void k_event_pool(const float* __restrict__ seq,
                             const int* __restrict__ eidx,
                             const float* __restrict__ emask,
                             bf16* __restrict__ evhi, bf16* __restrict__ evlo) {
  const int e   = blockIdx.x;
  const int tid = threadIdx.x;
  int   idx[KE];
  float mk[KE];
  float msum = 0.f;
  #pragma unroll
  for (int k = 0; k < KE; ++k) {
    idx[k] = eidx[e * KE + k];
    mk[k]  = emask[e * KE + k];
    msum  += mk[k];
  }
  const float inv = 1.0f / fmaxf(msum, 1.0f);
  for (int c = tid; c < H; c += blockDim.x) {
    float s = 0.f;
    #pragma unroll
    for (int k = 0; k < KE; ++k) s += seq[idx[k] * H + c] * mk[k];
    const float v = s * inv;
    bf16 hi, lo; split_bf16(v, hi, lo);
    evhi[e * H + c] = hi;
    evlo[e * H + c] = lo;
  }
}

// ---- W1 (1536x768 row-major) -> W1^T hi/lo (768x1536, K-contiguous) ----
__global__ void k_w1_transpose(const float* __restrict__ W1,
                               bf16* __restrict__ th, bf16* __restrict__ tl) {
  const int id = blockIdx.x * blockDim.x + threadIdx.x;   // < 1536*768
  const int k = id / H, n = id % H;
  const float x = W1[id];
  bf16 hi, lo; split_bf16(x, hi, lo);
  th[n * TWOH + k] = hi;
  tl[n * TWOH + k] = lo;
}

// ---- root branch folded to constant 2-vector: c = mean(ph[root])·Wb[768:] + bb ----
__global__ void k_root(const float* __restrict__ seq,
                       const int* __restrict__ pidx, const float* __restrict__ pmask,
                       const int* __restrict__ ridx,
                       const float* __restrict__ Wb, const float* __restrict__ bb,
                       float* __restrict__ cws) {
  const int tid = threadIdx.x;
  float c0 = 0.f, c1 = 0.f;
  for (int c = tid; c < H; c += 256) {
    float v = 0.f;
    for (int r = 0; r < NR; ++r) {
      const int p = ridx[r];
      float s = 0.f, ms = 0.f;
      #pragma unroll
      for (int k = 0; k < KP; ++k) {
        const int t = pidx[p * KP + k];
        const float m = pmask[p * KP + k];
        s  += seq[t * H + c] * m;
        ms += m;
      }
      v += s / fmaxf(ms, 1.0f);
    }
    v *= (1.0f / NR);
    c0 += v * Wb[(H + c) * 2 + 0];
    c1 += v * Wb[(H + c) * 2 + 1];
  }
  __shared__ float r0[256], r1[256];
  r0[tid] = c0; r1[tid] = c1;
  __syncthreads();
  for (int s = 128; s > 0; s >>= 1) {
    if (tid < s) { r0[tid] += r0[tid + s]; r1[tid] += r1[tid + s]; }
    __syncthreads();
  }
  if (tid == 0) { cws[0] = r0[0] + bb[0]; cws[1] = r1[0] + bb[1]; }
}

// ---- fused gathered GEMM + relu + Wb-dot epilogue ----
// A[m,k] = ev[pair[m, k<768?0:1]][k%768] (hi/lo bf16), B = W1 (via W1^T hi/lo),
// D = A x B split-product, then logits[m] = relu(D+b1)·Wb[:768] + c.
// B tile staged double-buffered via TDM tensor_load_to_lds (TENSORcnt), else
// GLOBAL_LOAD_ASYNC_TO_LDS (ASYNCcnt), else plain sync staging.
__global__ void __launch_bounds__(128)
k_pair_gemm(const bf16* __restrict__ evhi, const bf16* __restrict__ evlo,
            const bf16* __restrict__ w1thi, const bf16* __restrict__ w1tlo,
            const int* __restrict__ pair, const float* __restrict__ b1,
            const float* __restrict__ Wb, const float* __restrict__ cws,
            float* __restrict__ out) {
  __shared__ __align__(16) bf16 Bh[2][128 * LDB];
  __shared__ __align__(16) bf16 Bl[2][128 * LDB];

  const int tid  = threadIdx.x;
  const int wave = tid >> 5;
  const int lane = tid & 31;
  const int lrow = lane & 15;
  const int hsel = (lane < 16) ? 0 : 8;   // A fragment K-offset (ISA half-lane layout)
  const int bq   = (lane < 16) ? 0 : 1;   // B fragment uint4 offset (K {0-7,16-23} vs {8-15,24-31})

  const int gm = blockIdx.x * MT + wave * 16 + lrow;
  const int i0 = pair[2 * gm + 0];
  const int i1 = pair[2 * gm + 1];
  const bf16* a0h = evhi + (size_t)i0 * H;
  const bf16* a0l = evlo + (size_t)i0 * H;
  const bf16* a1h = evhi + (size_t)i1 * H;
  const bf16* a1l = evlo + (size_t)i1 * H;

  float s0[8], s1[8];
  #pragma unroll
  for (int r = 0; r < 8; ++r) { s0[r] = 0.f; s1[r] = 0.f; }

  const v8f vzero = {0.f, 0.f, 0.f, 0.f, 0.f, 0.f, 0.f, 0.f};

  // stage(bn, kk) into buffer (kk & 1)
  auto stage = [&](int sbn, int skk) {
    const int sbuf = skk & 1;
    const int kcol = skk * 32;
#if defined(HAVE_TDM)
    if (wave == 0) {   // one wave drives the DMA engine; 2 tensor ops per stage
      tdm_load_tile_128x64B(w1thi + (size_t)sbn * 128 * TWOH + kcol,
                            lds_offset(&Bh[sbuf][0]));
      tdm_load_tile_128x64B(w1tlo + (size_t)sbn * 128 * TWOH + kcol,
                            lds_offset(&Bl[sbuf][0]));
    }
#else
    #pragma unroll
    for (int c = 0; c < 4; ++c) {
      const int chunk = tid + c * 128;        // [0, 512) 16B-chunks per buffer
      const int n = chunk >> 2, j = chunk & 3;
      const bf16* srch = w1thi + (size_t)(sbn * 128 + n) * TWOH + kcol + j * 8;
      const bf16* srcl = w1tlo + (size_t)(sbn * 128 + n) * TWOH + kcol + j * 8;
      async_copy16(srch, &Bh[sbuf][n * LDB + j * 8]);
      async_copy16(srcl, &Bl[sbuf][n * LDB + j * 8]);
    }
#endif
  };

  stage(0, 0);   // prologue prefetch

  for (int bn = 0; bn < NBN; ++bn) {
    v8f acc[NSUB];
    #pragma unroll
    for (int t = 0; t < NSUB; ++t) acc[t] = vzero;

    for (int kk = 0; kk < KSTEPS; ++kk) {
      const int buf  = kk & 1;
      const int kcol = kk * 32;

      // issue prefetch of next tile into the other buffer, then wait for this one
      const bool has_next = (kk + 1 < KSTEPS) || (bn + 1 < NBN);
      if (has_next) {
        const int nkk = (kk + 1 < KSTEPS) ? kk + 1 : 0;
        const int nbn = (kk + 1 < KSTEPS) ? bn : bn + 1;
        stage(nbn, nkk);
        wait_stage(true);    // in-order completion => current buffer has landed
      } else {
        wait_stage(false);
      }
      __syncthreads();       // all waves observe the staged tile

      // A fragments: gathered ev rows, straight from global (L2 resident)
      FragAB ah, al;
      {
        const bf16* rh = (kk < 24) ? a0h : a1h;
        const bf16* rl = (kk < 24) ? a0l : a1l;
        const int  c2 = (kk < 24) ? kcol : (kcol - H);
        const uint4* ph = (const uint4*)(rh + c2 + hsel);
        const uint4* pl = (const uint4*)(rl + c2 + hsel);
        ah.q[0] = ph[0]; ah.q[1] = ph[2];   // K {0-7,16-23} / {8-15,24-31}
        al.q[0] = pl[0]; al.q[1] = pl[2];
      }

      #pragma unroll
      for (int t = 0; t < NSUB; ++t) {
        FragAB bh, bl;
        const uint4* pb = (const uint4*)(&Bh[buf][(t * 16 + lrow) * LDB]);
        const uint4* qb = (const uint4*)(&Bl[buf][(t * 16 + lrow) * LDB]);
        bh.q[0] = pb[bq]; bh.q[1] = pb[bq + 2];
        bl.q[0] = qb[bq]; bl.q[1] = qb[bq + 2];
        // split-precision product: Ah*Bh + Ah*Bl + Al*Bh, fp32 accumulate
        acc[t] = __builtin_amdgcn_wmma_f32_16x16x32_bf16(false, ah.v, false, bh.v,
                                                         (short)0, acc[t], false, false);
        acc[t] = __builtin_amdgcn_wmma_f32_16x16x32_bf16(false, ah.v, false, bl.v,
                                                         (short)0, acc[t], false, false);
        acc[t] = __builtin_amdgcn_wmma_f32_16x16x32_bf16(false, al.v, false, bh.v,
                                                         (short)0, acc[t], false, false);
      }

      __syncthreads();      // all reads of this buffer done before it is re-staged
    }

    // epilogue for this 128-wide N stripe: relu(acc + b1) · Wb columns
    #pragma unroll
    for (int t = 0; t < NSUB; ++t) {
      const int n = bn * 128 + t * 16 + lrow;
      const float b1v = b1[n];
      const float w0  = Wb[2 * n + 0];
      const float w1  = Wb[2 * n + 1];
      #pragma unroll
      for (int r = 0; r < 8; ++r) {
        float x = acc[t][r] + b1v;
        x = fmaxf(x, 0.f);
        s0[r] += x * w0;
        s1[r] += x * w1;
      }
    }
  }

  // C-layout: VGPR r holds M=r (lanes 0-15) / M=r+8 (lanes 16-31), N=lane&15.
  // Reduce the 16 lanes sharing a row, add constant root term, store logits.
  #pragma unroll
  for (int r = 0; r < 8; ++r) {
    float a = s0[r], b = s1[r];
    #pragma unroll
    for (int off = 1; off < 16; off <<= 1) {
      a += __shfl_xor(a, off, 32);
      b += __shfl_xor(b, off, 32);
    }
    if (lrow == 0) {
      const int m = blockIdx.x * MT + wave * 16 + r + ((lane < 16) ? 0 : 8);
      out[2 * m + 0] = a + cws[0];
      out[2 * m + 1] = b + cws[1];
    }
  }
}

extern "C" void kernel_launch(void* const* d_in, const int* in_sizes, int n_in,
                              void* d_out, int out_size, void* d_ws, size_t ws_size,
                              hipStream_t stream) {
  const float* seq         = (const float*)d_in[0];
  const int*   phrase_idx  = (const int*)  d_in[1];
  const float* phrase_mask = (const float*)d_in[2];
  const int*   event_idx   = (const int*)  d_in[3];
  const float* event_mask  = (const float*)d_in[4];
  const int*   pair_idx    = (const int*)  d_in[5];
  const int*   root_idx    = (const int*)  d_in[6];
  const float* W1          = (const float*)d_in[7];
  const float* b1          = (const float*)d_in[8];
  const float* Wb          = (const float*)d_in[9];
  const float* bb          = (const float*)d_in[10];
  float* out = (float*)d_out;

  char* ws = (char*)d_ws;
  bf16* evhi  = (bf16*)ws; ws += (size_t)EVn * H * sizeof(bf16);
  bf16* evlo  = (bf16*)ws; ws += (size_t)EVn * H * sizeof(bf16);
  bf16* w1thi = (bf16*)ws; ws += (size_t)H * TWOH * sizeof(bf16);
  bf16* w1tlo = (bf16*)ws; ws += (size_t)H * TWOH * sizeof(bf16);
  float* cws  = (float*)ws;

  k_event_pool<<<EVn, 256, 0, stream>>>(seq, event_idx, event_mask, evhi, evlo);
  k_w1_transpose<<<(TWOH * H) / 256, 256, 0, stream>>>(W1, w1thi, w1tlo);
  k_root<<<1, 256, 0, stream>>>(seq, phrase_idx, phrase_mask, root_idx, Wb, bb, cws);
  k_pair_gemm<<<En / MT, 128, 0, stream>>>(evhi, evlo, w1thi, w1tlo,
                                           pair_idx, b1, Wb, cws, out);
}
